// DecoderBlock_65798898975240
// MI455X (gfx1250) — compile-verified, matrix operand order checked by reading, not run on hardware
//
#include <hip/hip_runtime.h>
#include <hip/hip_bf16.h>

typedef __attribute__((ext_vector_type(16))) __bf16 v16bf;
typedef __attribute__((ext_vector_type(8)))  float  v8f;
typedef __attribute__((ext_vector_type(4)))  unsigned int v4u;
typedef __attribute__((ext_vector_type(8)))  int    v8i;
typedef __attribute__((ext_vector_type(4)))  int    v4i;

#if defined(__AMDGCN__) && __has_builtin(__builtin_amdgcn_tensor_load_to_lds)
#define HAVE_TDM 1
#else
#define HAVE_TDM 0
#endif

union FragU {
    uint4 q[2];
    v16bf v;
};

__device__ __forceinline__ unsigned short f2bf(float f) {
    unsigned int u = __builtin_bit_cast(unsigned int, f);
    unsigned int r = (u + 0x7FFFu + ((u >> 16) & 1u)) >> 16;  // round-to-nearest-even
    return (unsigned short)r;
}
__device__ __forceinline__ float bf2f(unsigned short h) {
    unsigned int u = ((unsigned int)h) << 16;
    return __builtin_bit_cast(float, u);
}

#if HAVE_TDM
// Issue one TDM 2-D tile load (bf16 elements) Global -> LDS.
// D# per CDNA5 ISA:
//  group0: count=1 | lds_addr | global_addr(57b) | type=2
//  group1: data_size=1(2B), tensor_dim0/1, tile_dim0/1, tensor_dim0_stride
// Tracked by TENSORcnt; same-wave TDM ops complete in order, so a
// double-buffered pipeline can wait "<=1 outstanding" for the older tile.
__device__ __forceinline__ void tdm_load_2d_bf16(
    unsigned lds_off, const unsigned short* gptr,
    unsigned tile_w, unsigned tile_h,
    unsigned tensor_w, unsigned tensor_h, unsigned row_stride) {
    unsigned long long ga = (unsigned long long)(const void*)gptr;
    v4u g0;
    g0[0] = 1u;                                   // count=1 (valid), no gather
    g0[1] = lds_off;                              // LDS byte address
    g0[2] = (unsigned)ga;                         // global_addr[31:0]
    g0[3] = (unsigned)((ga >> 32) & 0x01FFFFFFull) | (2u << 30);  // [56:32] | type=2
    v8i g1;
    g1[0] = (int)(1u << 16);                      // data_size=1 -> 2 bytes
    g1[1] = (int)((tensor_w & 0xFFFFu) << 16);    // tensor_dim0[15:0]
    g1[2] = (int)((tensor_w >> 16) | ((tensor_h & 0xFFFFu) << 16));
    g1[3] = (int)((tensor_h >> 16) | (tile_w << 16));  // tile_dim0
    g1[4] = (int)(tile_h & 0xFFFFu);              // tile_dim1 (tile_dim2=0)
    g1[5] = (int)row_stride;                      // tensor_dim0_stride[31:0]
    g1[6] = 0;
    g1[7] = 0;
    v4i z4 = {0, 0, 0, 0};
#if defined(__clang_major__) && __clang_major__ >= 23
    v8i z8 = {0, 0, 0, 0, 0, 0, 0, 0};
    __builtin_amdgcn_tensor_load_to_lds(g0, g1, z4, z4, z8, 0);
#else
    __builtin_amdgcn_tensor_load_to_lds(g0, g1, z4, z4, 0);
#endif
}
#endif  // HAVE_TDM

// -------------------------------------------------------------------------
// LayerNorm over E=1024: one block per token; writes f32 (for residual) and
// bf16 (as WMMA GEMM A-operand).
// -------------------------------------------------------------------------
__global__ __launch_bounds__(256) void ln_kernel(
    const float* __restrict__ x, const float* __restrict__ gamma,
    const float* __restrict__ beta, float* __restrict__ hf,
    unsigned short* __restrict__ hb, int E) {
    __shared__ float red[256];
    __shared__ float red2[256];
    __shared__ float s_mean, s_rstd;
    int row = blockIdx.x;
    int t = threadIdx.x;
    const float* xr = x + (size_t)row * E;
    float sum = 0.f, sq = 0.f;
    for (int i = t; i < E; i += 256) { float v = xr[i]; sum += v; sq += v * v; }
    red[t] = sum; red2[t] = sq;
    __syncthreads();
    for (int s = 128; s > 0; s >>= 1) {
        if (t < s) { red[t] += red[t + s]; red2[t] += red2[t + s]; }
        __syncthreads();
    }
    if (t == 0) {
        float m = red[0] / (float)E;
        float var = red2[0] / (float)E - m * m;
        s_mean = m;
        s_rstd = rsqrtf(var + 1e-5f);
    }
    __syncthreads();
    float m = s_mean, r = s_rstd;
    for (int i = t; i < E; i += 256) {
        float v = (xr[i] - m) * r * gamma[i] + beta[i];
        hf[(size_t)row * E + i] = v;
        hb[(size_t)row * E + i] = f2bf(v);
    }
}

// -------------------------------------------------------------------------
// Repack wq/wk/wv [H,E,hd] f32 -> fused bf16 B matrix [E][3*E] row-major.
// -------------------------------------------------------------------------
__global__ __launch_bounds__(256) void pack_qkv_kernel(
    const float* __restrict__ wq, const float* __restrict__ wk,
    const float* __restrict__ wv, unsigned short* __restrict__ out) {
    const int E = 1024, hd = 64;
    size_t id = (size_t)blockIdx.x * blockDim.x + threadIdx.x;
    size_t total = (size_t)E * 3 * E;
    if (id >= total) return;
    int col = (int)(id % (3 * E));
    int e   = (int)(id / (3 * E));
    int sel = col / E;
    int c   = col % E;
    int h = c / hd, kk = c % hd;
    const float* w = (sel == 0) ? wq : ((sel == 1) ? wk : wv);
    out[id] = f2bf(w[((size_t)h * E + e) * hd + kk]);
}

__global__ __launch_bounds__(256) void f32_to_bf16_kernel(
    const float* __restrict__ in, unsigned short* __restrict__ out, size_t n) {
    size_t id = (size_t)blockIdx.x * blockDim.x + threadIdx.x;
    if (id < n) out[id] = f2bf(in[id]);
}

// -------------------------------------------------------------------------
// WMMA bf16 GEMM: C[M,N] = A[M,K] * B[K,N] (+bias)(+resid)(ReLU?)
// Block tile 128x64, BK=32; 8 waves as 4x2 grid of 32x32 wave tiles:
// per K-step each wave loads 2 A-frags + 2 B-frags (contiguous 16B ds reads
// per the 16-bit operand layout) and issues 4 v_wmma_f32_16x16x32_bf16.
// A tiles are double-buffered and DMA'd Global->LDS by the Tensor Data
// Mover: tile k+1 is issued at the top of step k and we wait
// s_wait_tensorcnt<=1, so the DMA overlaps B staging + all 4 WMMAs.
// -------------------------------------------------------------------------
template <bool RELU, bool BF16OUT>
__global__ __launch_bounds__(256) void gemm_kernel(
    const unsigned short* __restrict__ A,   // [M][K] bf16
    const unsigned short* __restrict__ Bm,  // [K][N] bf16
    const float* __restrict__ bias,         // [N] or nullptr
    const float* __restrict__ resid,        // [M][N] f32 or nullptr
    void* __restrict__ outp,                // bf16 or f32 [M][N]
    int M, int N, int K) {
    constexpr int BM = 128, BN = 64, BK = 32;
    __shared__ unsigned short Asm[2][BM * BK];  // [m][k], double-buffered (TDM)
    __shared__ unsigned short BsmT[BN * BK];    // [n][k]
    int n0 = blockIdx.x * BN;
    int m0 = blockIdx.y * BM;
    int t = threadIdx.x;
    int wave = t >> 5, lane = t & 31;
    int lm = lane & 15, lhi = lane >> 4;
    int mi0 = (wave >> 1) * 32;   // wave row tile
    int ni0 = (wave & 1) * 32;    // wave col tile
    int nk = K / BK;

    v8f acc[2][2];
#pragma unroll
    for (int a = 0; a < 2; a++)
#pragma unroll
        for (int b = 0; b < 2; b++)
#pragma unroll
            for (int e = 0; e < 8; e++) acc[a][b][e] = 0.f;

#if HAVE_TDM
    if (t < 32)   // prologue: DMA tile 0 into buffer 0
        tdm_load_2d_bf16((unsigned)(size_t)(void*)Asm[0],
                         A + (size_t)m0 * K,
                         BK, BM, (unsigned)K, (unsigned)M, (unsigned)K);
#endif

    for (int ki = 0; ki < nk; ki++) {
        int k0 = ki * BK;
#if HAVE_TDM
        if (t < 32 && ki + 1 < nk)   // DMA next tile into the other buffer
            tdm_load_2d_bf16((unsigned)(size_t)(void*)Asm[(ki + 1) & 1],
                             A + (size_t)m0 * K + (k0 + BK),
                             BK, BM, (unsigned)K, (unsigned)M, (unsigned)K);
        const unsigned short* Abuf = Asm[ki & 1];
#else
        {   // fallback: manual A staging (single buffer)
            int row = t >> 1, half = t & 1;
            const uint4* gp = reinterpret_cast<const uint4*>(
                A + (size_t)(m0 + row) * K + k0 + half * 16);
            uint4* sp = reinterpret_cast<uint4*>(&Asm[0][row * BK + half * 16]);
            sp[0] = gp[0];
            sp[1] = gp[1];
        }
        const unsigned short* Abuf = Asm[0];
#endif
        {   // B tile: 32 k-rows x 64 cols, stored transposed [n][k]
            int k = t >> 3, nch = (t & 7) * 8;
            const unsigned short* gp = Bm + (size_t)(k0 + k) * N + n0 + nch;
#pragma unroll
            for (int i = 0; i < 8; i++) BsmT[(nch + i) * BK + k] = gp[i];
            if (k0 + BK < K)  // pull next B tile toward L2/L0
                __builtin_prefetch(Bm + (size_t)(k0 + BK + k) * N + n0 + nch, 0, 1);
        }
#if HAVE_TDM
        if (t < 32) {   // drain oldest TDM load (newest may stay in flight)
            if (ki + 1 < nk) __builtin_amdgcn_s_wait_tensorcnt(1);
            else             __builtin_amdgcn_s_wait_tensorcnt(0);
        }
#endif
        __syncthreads();
        FragU af[2], bfr[2];
#pragma unroll
        for (int a = 0; a < 2; a++) {
            const unsigned short* p = &Abuf[(mi0 + a * 16 + lm) * BK + lhi * 8];
            af[a].q[0] = *reinterpret_cast<const uint4*>(p);
            af[a].q[1] = *reinterpret_cast<const uint4*>(p + 16);
        }
#pragma unroll
        for (int b = 0; b < 2; b++) {
            const unsigned short* p = &BsmT[(ni0 + b * 16 + lm) * BK + lhi * 8];
            bfr[b].q[0] = *reinterpret_cast<const uint4*>(p);
            bfr[b].q[1] = *reinterpret_cast<const uint4*>(p + 16);
        }
#pragma unroll
        for (int a = 0; a < 2; a++)
#pragma unroll
            for (int b = 0; b < 2; b++)
                acc[a][b] = __builtin_amdgcn_wmma_f32_16x16x32_bf16(
                    false, af[a].v, false, bfr[b].v, (short)0, acc[a][b], false, false);
        __syncthreads();
    }
    // C/D layout: vgpr r holds M=r (lanes 0-15) / M=r+8 (lanes 16-31), N=lane&15
#pragma unroll
    for (int a = 0; a < 2; a++)
#pragma unroll
        for (int b = 0; b < 2; b++)
#pragma unroll
            for (int r = 0; r < 8; r++) {
                int row = m0 + mi0 + a * 16 + r + (lhi << 3);
                int col = n0 + ni0 + b * 16 + lm;
                float v = acc[a][b][r];
                if (bias)  v += bias[col];
                if (resid) v += resid[(size_t)row * N + col];
                if (RELU)  v = v > 0.f ? v : 0.f;
                if (BF16OUT)
                    ((unsigned short*)outp)[(size_t)row * N + col] = f2bf(v);
                else
                    ((float*)outp)[(size_t)row * N + col] = v;
            }
}

// -------------------------------------------------------------------------
// Flash attention: one block per (b,h, 64-query tile). 8 waves; wave owns a
// 16x32 piece of the 64x64 S / O tiles. S = (Q*scale)K^T and O += P V via
// WMMA; causal online softmax in LDS. K tiles are double-buffered TDM DMAs:
// tile j+1 is issued after the post-S barrier of step j, overlapping the
// softmax and P.V phases.
// -------------------------------------------------------------------------
__global__ __launch_bounds__(256) void attn_kernel(
    const unsigned short* __restrict__ qkv,  // [B*T][3E] bf16 (q|k|v)
    unsigned short* __restrict__ o,          // [B*T][E] bf16 (head-concat)
    int B, int T, int H, int hd) {
    const int E = H * hd;   // 1024
    const int W3 = 3 * E;   // 3072
    __shared__ unsigned short Qs[64 * 64];     // [m][k] scaled bf16
    __shared__ unsigned short Ks[2][64 * 64];  // [n][k], double-buffered (TDM)
    __shared__ unsigned short VsT[64 * 64];    // [n=hd][k=key]
    __shared__ float          Sf[64 * 64];     // scores f32
    __shared__ unsigned short Ps[64 * 64];     // exp probs bf16
    __shared__ float mrow[64], lrow[64], arow[64];

    int qtile = blockIdx.x;
    int bh = blockIdx.y;
    int b = bh / H, h = bh % H;
    int q0 = qtile * 64;
    int t = threadIdx.x;
    int wave = t >> 5, lane = t & 31;
    int lm = lane & 15, lhi = lane >> 4;
    int mi = wave >> 1;          // 0..3 (16-row tile)
    int nbase = (wave & 1) * 2;  // col tiles {nbase, nbase+1}

    const float scale = rsqrtf((float)hd);

    {   // Q tile, pre-scaled
        int m = t >> 2, c0 = (t & 3) * 16;
        const unsigned short* gp = qkv + (size_t)(b * T + q0 + m) * W3 + h * hd + c0;
#pragma unroll
        for (int i = 0; i < 16; i++) Qs[m * 64 + c0 + i] = f2bf(bf2f(gp[i]) * scale);
    }
    if (t < 64) { mrow[t] = -3.0e38f; lrow[t] = 0.f; }

    v8f oacc[2];
#pragma unroll
    for (int i = 0; i < 2; i++)
#pragma unroll
        for (int e = 0; e < 8; e++) oacc[i][e] = 0.f;

#if HAVE_TDM
    if (t < 32)   // prologue: DMA K tile 0
        tdm_load_2d_bf16((unsigned)(size_t)(void*)Ks[0],
                         qkv + (size_t)(b * T) * W3 + E + h * hd,
                         64, 64, (unsigned)W3, (unsigned)(B * T), (unsigned)W3);
#endif

    for (int j = 0; j <= qtile; j++) {
        int j0 = j * 64;
#if !HAVE_TDM
        {   // fallback: manual K staging
            int n = t >> 2, c0 = (t & 3) * 16;
            const uint4* gp = reinterpret_cast<const uint4*>(
                qkv + (size_t)(b * T + j0 + n) * W3 + E + h * hd + c0);
            uint4* sp = reinterpret_cast<uint4*>(&Ks[0][n * 64 + c0]);
            sp[0] = gp[0];
            sp[1] = gp[1];
        }
#endif
        {   // V transposed: [n=hd col][k=key row]
            int s = t >> 2, c0 = (t & 3) * 16;
            const unsigned short* gp = qkv + (size_t)(b * T + j0 + s) * W3 + 2 * E + h * hd + c0;
#pragma unroll
            for (int i = 0; i < 16; i++) VsT[(c0 + i) * 64 + s] = gp[i];
        }
#if HAVE_TDM
        if (t < 32) __builtin_amdgcn_s_wait_tensorcnt(0);  // K tile j landed
        const unsigned short* Kbuf = Ks[j & 1];
#else
        const unsigned short* Kbuf = Ks[0];
#endif
        __syncthreads();
        // S = Q K^T for this wave's two 16x16 tiles (K dim = 64 -> 2 wmma each)
#pragma unroll
        for (int nn = 0; nn < 2; nn++) {
            int ni = nbase + nn;
            v8f sacc;
#pragma unroll
            for (int e = 0; e < 8; e++) sacc[e] = 0.f;
#pragma unroll
            for (int ks = 0; ks < 2; ks++) {
                FragU af, bfr;
                af.q[0] = *reinterpret_cast<const uint4*>(&Qs[(mi * 16 + lm) * 64 + ks * 32 + lhi * 8]);
                af.q[1] = *reinterpret_cast<const uint4*>(&Qs[(mi * 16 + lm) * 64 + ks * 32 + 16 + lhi * 8]);
                bfr.q[0] = *reinterpret_cast<const uint4*>(&Kbuf[(ni * 16 + lm) * 64 + ks * 32 + lhi * 8]);
                bfr.q[1] = *reinterpret_cast<const uint4*>(&Kbuf[(ni * 16 + lm) * 64 + ks * 32 + 16 + lhi * 8]);
                sacc = __builtin_amdgcn_wmma_f32_16x16x32_bf16(
                    false, af.v, false, bfr.v, (short)0, sacc, false, false);
            }
#pragma unroll
            for (int r = 0; r < 8; r++)
                Sf[(mi * 16 + r + (lhi << 3)) * 64 + ni * 16 + lm] = sacc[r];
        }
        __syncthreads();
#if HAVE_TDM
        if (t < 32 && j + 1 <= qtile)   // overlap K tile j+1 with softmax + PV
            tdm_load_2d_bf16((unsigned)(size_t)(void*)Ks[(j + 1) & 1],
                             qkv + (size_t)(b * T + j0 + 64) * W3 + E + h * hd,
                             64, 64, (unsigned)W3, (unsigned)(B * T), (unsigned)W3);
#endif
        // online softmax with causal mask: one thread per row
        if (t < 64) {
            int row = t;
            int qidx = q0 + row;
            float mprev = mrow[row];
            float mx = mprev;
            for (int c = 0; c < 64; c++) {
                int key = j0 + c;
                float s = (key <= qidx) ? Sf[row * 64 + c] : -3.0e38f;
                mx = fmaxf(mx, s);
            }
            float alpha = __expf(mprev - mx);
            float sum = 0.f;
            for (int c = 0; c < 64; c++) {
                int key = j0 + c;
                float p = (key <= qidx) ? __expf(Sf[row * 64 + c] - mx) : 0.f;
                sum += p;
                Ps[row * 64 + c] = f2bf(p);
            }
            mrow[row] = mx;
            lrow[row] = lrow[row] * alpha + sum;
            arow[row] = alpha;
        }
        __syncthreads();
        // rescale running O, then O += P V
#pragma unroll
        for (int nn = 0; nn < 2; nn++) {
#pragma unroll
            for (int r = 0; r < 8; r++) {
                int row = mi * 16 + r + (lhi << 3);
                oacc[nn][r] *= arow[row];
            }
            int ni = nbase + nn;
#pragma unroll
            for (int ks = 0; ks < 2; ks++) {
                FragU af, bfr;
                af.q[0] = *reinterpret_cast<const uint4*>(&Ps[(mi * 16 + lm) * 64 + ks * 32 + lhi * 8]);
                af.q[1] = *reinterpret_cast<const uint4*>(&Ps[(mi * 16 + lm) * 64 + ks * 32 + 16 + lhi * 8]);
                bfr.q[0] = *reinterpret_cast<const uint4*>(&VsT[(ni * 16 + lm) * 64 + ks * 32 + lhi * 8]);
                bfr.q[1] = *reinterpret_cast<const uint4*>(&VsT[(ni * 16 + lm) * 64 + ks * 32 + 16 + lhi * 8]);
                oacc[nn] = __builtin_amdgcn_wmma_f32_16x16x32_bf16(
                    false, af.v, false, bfr.v, (short)0, oacc[nn], false, false);
            }
        }
        __syncthreads();
    }
    // normalize and write head-concat output (bf16, feeds proj GEMM)
#pragma unroll
    for (int nn = 0; nn < 2; nn++) {
        int ni = nbase + nn;
#pragma unroll
        for (int r = 0; r < 8; r++) {
            int row = mi * 16 + r + (lhi << 3);
            int col = ni * 16 + lm;
            float v = oacc[nn][r] / lrow[row];
            o[(size_t)(b * T + q0 + row) * E + h * hd + col] = f2bf(v);
        }
    }
}

// -------------------------------------------------------------------------
extern "C" void kernel_launch(void* const* d_in, const int* in_sizes, int n_in,
                              void* d_out, int out_size, void* d_ws, size_t ws_size,
                              hipStream_t stream) {
    const int Bsz = 2, T = 2048, E = 1024, H = 16, hd = 64, dff = 4096;
    const int M = Bsz * T;  // 4096 tokens

    const float* x      = (const float*)d_in[0];
    const float* wq     = (const float*)d_in[1];
    const float* wk     = (const float*)d_in[2];
    const float* wv     = (const float*)d_in[3];
    const float* w_proj = (const float*)d_in[4];
    const float* b_proj = (const float*)d_in[5];
    const float* gamma1 = (const float*)d_in[6];
    const float* beta1  = (const float*)d_in[7];
    const float* gamma2 = (const float*)d_in[8];
    const float* beta2  = (const float*)d_in[9];
    const float* w1     = (const float*)d_in[10];
    const float* b1     = (const float*)d_in[11];
    const float* w2     = (const float*)d_in[12];
    const float* b2     = (const float*)d_in[13];
    float* out = (float*)d_out;

    // Workspace layout (bytes); later stages reuse regions that are dead.
    char* ws = (char*)d_ws;
    const size_t MB = 1024ull * 1024ull;
    float*          hf    = (float*)(ws + 0);              // 16MB  LN1 f32 (residual)
    unsigned short* hb    = (unsigned short*)(ws + 16*MB); //  8MB  LN1 bf16
    unsigned short* wqkvb = (unsigned short*)(ws + 24*MB); //  6MB  fused qkv weights bf16
    unsigned short* qkv   = (unsigned short*)(ws + 30*MB); // 24MB  q|k|v bf16 [4096][3072]
    unsigned short* ob    = (unsigned short*)(ws + 54*MB); //  8MB  attn out bf16
    unsigned short* wpb   = (unsigned short*)(ws + 62*MB); //  2MB  w_proj bf16
    float*          x2    = (float*)(ws + 16*MB);          // 16MB  reuses hb+wqkvb (dead)
    float*          h2    = (float*)(ws + 32*MB);          // 16MB  reuses qkv (dead)
    unsigned short* h2b   = (unsigned short*)(ws + 48*MB); //  8MB  reuses qkv tail/ob
    unsigned short* w1b   = (unsigned short*)(ws + 56*MB); //  8MB  reuses ob tail/wpb
    unsigned short* t1    = (unsigned short*)(ws + 64*MB); // 32MB  relu(h2 w1 + b1) bf16
    unsigned short* w2b   = (unsigned short*)(ws + 96*MB); //  8MB  w2 bf16
    // peak footprint: 104MB

    dim3 blk(256);

    // 1) LN1
    ln_kernel<<<M, blk, 0, stream>>>(x, gamma1, beta1, hf, hb, E);

    // 2) fuse/convert qkv weights, then one [4096,1024]x[1024,3072] WMMA GEMM
    {
        unsigned int nb = (unsigned int)(((size_t)E * 3 * E + 255) / 256);
        pack_qkv_kernel<<<nb, blk, 0, stream>>>(wq, wk, wv, wqkvb);
    }
    gemm_kernel<false, true><<<dim3(3 * E / 64, M / 128), blk, 0, stream>>>(
        hb, wqkvb, nullptr, nullptr, qkv, M, 3 * E, E);

    // 3) flash attention (causal), bf16 head-concat output
    attn_kernel<<<dim3(T / 64, Bsz * H), blk, 0, stream>>>(qkv, ob, Bsz, T, H, hd);

    // 4) output projection + bias + residual(h) -> x2 (f32)
    f32_to_bf16_kernel<<<(unsigned int)(((size_t)E * E + 255) / 256), blk, 0, stream>>>(
        w_proj, wpb, (size_t)E * E);
    gemm_kernel<false, false><<<dim3(E / 64, M / 128), blk, 0, stream>>>(
        ob, wpb, b_proj, hf, x2, M, E, E);

    // 5) LN2
    ln_kernel<<<M, blk, 0, stream>>>(x2, gamma2, beta2, h2, h2b, E);

    // 6) FFN1: relu(h2 w1 + b1) -> bf16
    f32_to_bf16_kernel<<<(unsigned int)(((size_t)E * dff + 255) / 256), blk, 0, stream>>>(
        w1, w1b, (size_t)E * dff);
    gemm_kernel<true, true><<<dim3(dff / 64, M / 128), blk, 0, stream>>>(
        h2b, w1b, b1, nullptr, t1, M, dff, E);

    // 7) FFN2: t1 w2 + b2 + residual(h2) -> d_out (f32)
    f32_to_bf16_kernel<<<(unsigned int)(((size_t)dff * E + 255) / 256), blk, 0, stream>>>(
        w2, w2b, (size_t)dff * E);
    gemm_kernel<false, false><<<dim3(E / 64, M / 128), blk, 0, stream>>>(
        t1, w2b, b2, h2, out, M, E, dff);
}
